// TensorNeuralNetwork_10625749090782
// MI455X (gfx1250) — compile-verified, matrix operand order checked by reading, not run on hardware
//
#include <hip/hip_runtime.h>

// Problem constants (match reference)
#define Bsz  4096
#define Rr   50
#define Dd   20
#define Hh   50
#define EPSc 1e-5f

// Tiling
#define ROWS 128      // batch rows per block (8 waves x 16-row WMMA tiles)
#define KPAD 64       // H padded to 2 x K32 WMMA steps
#define NPAD 64       // H padded to 4 x N16 tiles
#define H1S  72       // h1 LDS row stride in halfs (16B aligned rows)
#define WTS  72       // Wt LDS row stride in halfs
#define H2S  65       // h2 LDS row stride in floats (bank-conflict-free column walks)

typedef __attribute__((ext_vector_type(16))) _Float16 v16h;
typedef __attribute__((ext_vector_type(8)))  _Float16 v8h;
typedef __attribute__((ext_vector_type(8)))  float    v8f;

__global__ void tnn_zero(float* __restrict__ out, int n) {
    int i = blockIdx.x * blockDim.x + threadIdx.x;
    if (i < n) out[i] = 0.0f;
}

__global__ __launch_bounds__(256) void tnn_fused(
    const float* __restrict__ x,   const float* __restrict__ W1,
    const float* __restrict__ b1,  const float* __restrict__ g1,
    const float* __restrict__ be1, const float* __restrict__ W2,
    const float* __restrict__ b2,  const float* __restrict__ g2,
    const float* __restrict__ be2, const float* __restrict__ W3,
    const float* __restrict__ b3,  const float* __restrict__ theta,
    float* __restrict__ out)
{
    __shared__ alignas(16) _Float16 h1s[ROWS * H1S];   // sin(LN(h1)) tiles, f16, K-padded
    __shared__ alignas(16) _Float16 Wts[NPAD * WTS];   // W2^T tile: Wts[n][k], f16, padded
    __shared__ float h2s[ROWS * H2S];                  // GEMM accumulators, f32
    __shared__ float W1s[Hh], b1s[Hh], g1s[Hh], be1s[Hh];
    __shared__ float b2s[Hh], g2s[Hh], be2s[Hh], W3s[Hh];
    __shared__ float b3sh;

    const int tid = threadIdx.x;
    const int r   = blockIdx.y;

    // Zero everything once: pad lanes of h1s/Wts must stay 0 so 0-padded K never
    // multiplies garbage (0*NaN would poison the accumulators).
    for (int i = tid; i < ROWS * H1S; i += 256) h1s[i] = (_Float16)0.0f;
    for (int i = tid; i < NPAD * WTS; i += 256) Wts[i] = (_Float16)0.0f;
    __syncthreads();

    float prod = 1.0f;   // product over d (threads < ROWS)
    float xd   = 0.0f;

    for (int d = 0; d < Dd; ++d) {
        // ---- phase A: stage per-(r,d) params + transposed f16 W2 into LDS ----
        const int pbase = (r * Dd + d) * Hh;
        if (tid < Hh) {
            W1s[tid]  = W1[pbase + tid];  b1s[tid] = b1[pbase + tid];
            g1s[tid]  = g1[pbase + tid];  be1s[tid] = be1[pbase + tid];
            b2s[tid]  = b2[pbase + tid];  g2s[tid] = g2[pbase + tid];
            be2s[tid] = be2[pbase + tid]; W3s[tid] = W3[pbase + tid];
        }
        if (tid == 0) b3sh = b3[r * Dd + d];
        const float* W2p = W2 + (size_t)(r * Dd + d) * Hh * Hh;   // [k_in][n_out]
        for (int i = tid; i < Hh * Hh; i += 256) {
            int k = i / Hh, n = i % Hh;
            Wts[n * WTS + k] = (_Float16)W2p[i];
        }
        // prefetch next d's W2 tile into cache while we compute this one
        if (d + 1 < Dd && tid * 32 < Hh * Hh) {
            const float* nxt = W2 + (size_t)(r * Dd + d + 1) * Hh * Hh;
            __builtin_prefetch(nxt + tid * 32, 0, 0);
        }
        __syncthreads();

        // ---- phase B: h1 = sin(LN(x_d*W1 + b1)) per row, stored f16 ----
        if (tid < ROWS) {
            const int gRow = blockIdx.x * ROWS + tid;
            xd = x[gRow * Dd + d];
            float s = 0.0f, q = 0.0f;
            for (int h = 0; h < Hh; ++h) {
                float v = fmaf(xd, W1s[h], b1s[h]);
                s += v; q += v * v;
            }
            float m   = s * (1.0f / Hh);
            float var = q * (1.0f / Hh) - m * m;
            float rs  = rsqrtf(var + EPSc);
            _Float16* hrow = &h1s[tid * H1S];
            for (int h = 0; h < Hh; ++h) {
                float v  = fmaf(xd, W1s[h], b1s[h]);
                float hn = __sinf(fmaf((v - m) * rs, g1s[h], be1s[h]));
                hrow[h] = (_Float16)hn;
            }
        }
        __syncthreads();

        // ---- phase C: h2 = h1 @ W2 via v_wmma_f32_16x16x32_f16 ----
        // All 256 threads; no divergence (EXEC must be all-1s for WMMA).
        {
            const int wave  = tid >> 5;
            const int lane  = tid & 31;
            const int sub   = lane >> 4;     // lane group selects K+0 / K+8 halves
            const int ml    = lane & 15;     // A: row in tile;  B: column (N)
            const int koff  = sub * 8;
            const int mBase = wave * 16;

            v8f acc[4] = {v8f{}, v8f{}, v8f{}, v8f{}};
            const _Float16* arow = &h1s[(mBase + ml) * H1S];
#pragma unroll
            for (int kb = 0; kb < KPAD; kb += 32) {
                v8h alo = *(const v8h*)(arow + kb + koff);
                v8h ahi = *(const v8h*)(arow + kb + 16 + koff);
                v16h a = __builtin_shufflevector(alo, ahi,
                         0,1,2,3,4,5,6,7,8,9,10,11,12,13,14,15);
#pragma unroll
                for (int nt = 0; nt < 4; ++nt) {
                    const _Float16* brow = &Wts[(nt * 16 + ml) * WTS];
                    v8h blo = *(const v8h*)(brow + kb + koff);
                    v8h bhi = *(const v8h*)(brow + kb + 16 + koff);
                    v16h bf16 = __builtin_shufflevector(blo, bhi,
                               0,1,2,3,4,5,6,7,8,9,10,11,12,13,14,15);
                    acc[nt] = __builtin_amdgcn_wmma_f32_16x16x32_f16(
                        /*neg_a=*/false, a, /*neg_b=*/false, bf16,
                        /*c_mod=*/(short)0, acc[nt],
                        /*reuse_a=*/false, /*reuse_b=*/false);
                }
            }
            // D layout: VGPR v -> M = v + 8*sub, lane%16 -> N
#pragma unroll
            for (int nt = 0; nt < 4; ++nt)
#pragma unroll
                for (int v = 0; v < 8; ++v)
                    h2s[(mBase + v + 8 * sub) * H2S + nt * 16 + ml] = acc[nt][v];
        }
        __syncthreads();

        // ---- phase D: LN + sin + W3 dot + sin + boundary factor + product ----
        if (tid < ROWS) {
            const float* hrow = &h2s[tid * H2S];
            float s = 0.0f, q = 0.0f;
            for (int k = 0; k < Hh; ++k) {
                float v = hrow[k] + b2s[k];
                s += v; q += v * v;
            }
            float m   = s * (1.0f / Hh);
            float var = q * (1.0f / Hh) - m * m;
            float rs  = rsqrtf(var + EPSc);
            float dot = 0.0f;
            for (int k = 0; k < Hh; ++k) {
                float v  = hrow[k] + b2s[k];
                float hn = __sinf(fmaf((v - m) * rs, g2s[k], be2s[k]));
                dot = fmaf(hn, W3s[k], dot);
            }
            float o    = __sinf(dot + b3sh);
            float bfac = 8.0f * xd * (1.0f - xd);   // 2*(x-a)*(b-x)/((b-a)/2)^2, a=0,b=1
            prod *= o * bfac;
        }
        __syncthreads();   // protect params/Wt/h2s before next iteration rewrites
    }

    if (tid < ROWS) {
        const int gRow = blockIdx.x * ROWS + tid;
        atomicAdd(&out[gRow], theta[r] * prod);
    }
}

extern "C" void kernel_launch(void* const* d_in, const int* in_sizes, int n_in,
                              void* d_out, int out_size, void* d_ws, size_t ws_size,
                              hipStream_t stream) {
    (void)in_sizes; (void)n_in; (void)d_ws; (void)ws_size; (void)out_size;
    const float* x     = (const float*)d_in[0];
    const float* W1    = (const float*)d_in[1];
    const float* b1    = (const float*)d_in[2];
    const float* g1    = (const float*)d_in[3];
    const float* be1   = (const float*)d_in[4];
    const float* W2    = (const float*)d_in[5];
    const float* b2    = (const float*)d_in[6];
    const float* g2    = (const float*)d_in[7];
    const float* be2   = (const float*)d_in[8];
    const float* W3    = (const float*)d_in[9];
    const float* b3    = (const float*)d_in[10];
    const float* theta = (const float*)d_in[11];
    float* out = (float*)d_out;

    tnn_zero<<<(Bsz + 255) / 256, 256, 0, stream>>>(out, Bsz);

    dim3 grid(Bsz / ROWS, Rr);
    tnn_fused<<<grid, 256, 0, stream>>>(x, W1, b1, g1, be1, W2, b2, g2, be2,
                                        W3, b3, theta, out);
}
